// DestationaryAttention_36094905156115
// MI455X (gfx1250) — matrix operand order repaired and checked by reading, not compiled, not run on hardware
//
#include <hip/hip_runtime.h>

#define IN      256
#define P       64
#define TILE_B  32
#define XPITCH  264   // 256 + 8 halves pad -> row stride 528B, de-conflicts banks
#define QPITCH  68
#define OPITCH  72

typedef __attribute__((ext_vector_type(16))) _Float16 v16h;
typedef __attribute__((ext_vector_type(8)))  _Float16 v8h;
typedef __attribute__((ext_vector_type(8)))  float    v8f;

__device__ __forceinline__ v16h pack16(v8h lo, v8h hi) {
  v16h r;
#pragma unroll
  for (int i = 0; i < 8; ++i) { r[i] = lo[i]; r[i + 8] = hi[i]; }
  return r;
}

// Pack Wq/Wk/Wv into one [192][256] f16 matrix, Wo into [256][64] f16.
__global__ void convert_weights_kernel(const float* __restrict__ Wq,
                                       const float* __restrict__ Wk,
                                       const float* __restrict__ Wv,
                                       const float* __restrict__ Wo,
                                       _Float16* __restrict__ Wqkv16,
                                       _Float16* __restrict__ Wo16) {
  int idx = blockIdx.x * blockDim.x + threadIdx.x;
  if (idx < 3 * P * IN) {
    int r = idx / IN;
    int c = idx - r * IN;
    float w = (r < P) ? Wq[r * IN + c]
            : (r < 2 * P) ? Wk[(r - P) * IN + c]
                          : Wv[(r - 2 * P) * IN + c];
    Wqkv16[idx] = (_Float16)w;
  }
  if (idx < IN * P) Wo16[idx] = (_Float16)Wo[idx];
}

__launch_bounds__(256)
__global__ void fused_attn_kernel(const float* __restrict__ x,
                                  const float* __restrict__ bq,
                                  const float* __restrict__ bk,
                                  const float* __restrict__ bv,
                                  const float* __restrict__ bo,
                                  const _Float16* __restrict__ Wqkv16,
                                  const _Float16* __restrict__ Wo16,
                                  float* __restrict__ out) {
  __shared__ _Float16 xtile[TILE_B][XPITCH];
  __shared__ float    qsh[TILE_B][QPITCH];
  __shared__ float    ksh[TILE_B][QPITCH];
  __shared__ float    vsh[TILE_B][QPITCH];
  __shared__ _Float16 otile[TILE_B][OPITCH];

  const int tid  = threadIdx.x;
  const int wave = tid >> 5;
  const int lane = tid & 31;
  const int lr   = lane & 15;   // M (A/C) or N (B/C) position
  const int lh   = lane >> 4;   // lane half: K/M offset selector
  const int b0   = blockIdx.x * TILE_B;

  // ---- Phase 0: cooperative x-tile load, f32 -> f16 into LDS -------------
  {
    const int colBlk = (tid & 63) * 4;   // 0..252 step 4
    const int rowOff = tid >> 6;         // 0..3
#pragma unroll
    for (int t = 0; t < 8; ++t) {
      const int row = t * 4 + rowOff;
      const float4 f = *(const float4*)(x + (size_t)(b0 + row) * IN + colBlk);
      xtile[row][colBlk + 0] = (_Float16)f.x;
      xtile[row][colBlk + 1] = (_Float16)f.y;
      xtile[row][colBlk + 2] = (_Float16)f.z;
      xtile[row][colBlk + 3] = (_Float16)f.w;
    }
  }
  __syncthreads();

  // ---- Phase 1: QKV = x @ W^T + b via v_wmma_f32_16x16x32_f16 ------------
  // 12 tasks: mat(3) x mtile(2) x nhalf(2); each task = 2 N-tiles x 8 K-steps.
  for (int task = wave; task < 12; task += 8) {
    const int mat   = task >> 2;
    const int mtile = (task >> 1) & 1;
    const int nh    = task & 1;
    const _Float16* Wb  = Wqkv16 + (size_t)mat * P * IN;
    const float* bias   = (mat == 0) ? bq : (mat == 1) ? bk : bv;
    float* dst          = (mat == 0) ? &qsh[0][0] : (mat == 1) ? &ksh[0][0] : &vsh[0][0];
#pragma unroll
    for (int nt2 = 0; nt2 < 2; ++nt2) {
      const int nt = nh * 2 + nt2;
      const int n  = nt * 16 + lr;
      v8f c = {};
#pragma unroll
      for (int kk = 0; kk < 8; ++kk) {
        // A: 16x32 f16 tile, lane=M (lr), per-lane K = kb..kb+7, kb+16..kb+23
        const int arow = mtile * 16 + lr;
        const int akb  = kk * 32 + lh * 8;
        v8h alo = *(const v8h*)&xtile[arow][akb];
        v8h ahi = *(const v8h*)&xtile[arow][akb + 16];
        // B: 32x16, lane=N, per-lane K = kb..kb+7, kb+16..kb+23 (row-major W[n][k])
        const _Float16* wr = Wb + (size_t)n * IN + kk * 32 + lh * 8;
        v8h blo = *(const v8h*)wr;
        v8h bhi = *(const v8h*)(wr + 16);
        c = __builtin_amdgcn_wmma_f32_16x16x32_f16(false, pack16(alo, ahi),
                                                   false, pack16(blo, bhi),
                                                   (short)0, c, false, false);
      }
      const float bn = bias[n];
#pragma unroll
      for (int r = 0; r < 8; ++r) {
        const int row = mtile * 16 + r + lh * 8;   // C layout: M = r (+8 upper half)
        dst[row * QPITCH + n] = c[r] + bn;
      }
    }
  }
  __syncthreads();

  // ---- Phase 2: per-sample rank-1 softmax-attention (f32 VALU) -----------
  {
    const float inv_s = 0.125f;  // 1/sqrt(64)
#pragma unroll
    for (int rr = 0; rr < TILE_B / 8; ++rr) {
      const int row = wave * (TILE_B / 8) + rr;
      const float q1 = qsh[row][lane] * inv_s;
      const float q2 = qsh[row][lane + 32] * inv_s;
      float kmax = fmaxf(ksh[row][lane], ksh[row][lane + 32]);
      float kmin = fminf(ksh[row][lane], ksh[row][lane + 32]);
#pragma unroll
      for (int o = 16; o >= 1; o >>= 1) {
        kmax = fmaxf(kmax, __shfl_xor(kmax, o, 32));
        kmin = fminf(kmin, __shfl_xor(kmin, o, 32));
      }
      // scores row i is q_i * k; its max is q_i*kmax or q_i*kmin by sign
      const float m1 = q1 * ((q1 >= 0.f) ? kmax : kmin);
      const float m2 = q2 * ((q2 >= 0.f) ? kmax : kmin);
      float n1 = 0.f, d1 = 0.f, n2 = 0.f, d2 = 0.f;
      for (int j = 0; j < P; ++j) {
        const float kj = ksh[row][j];   // LDS broadcast (all lanes same addr)
        const float vj = vsh[row][j];
        const float e1 = __expf(__builtin_fmaf(q1, kj, -m1));
        const float e2 = __expf(__builtin_fmaf(q2, kj, -m2));
        n1 = __builtin_fmaf(e1, vj, n1); d1 += e1;
        n2 = __builtin_fmaf(e2, vj, n2); d2 += e2;
      }
      otile[row][lane]      = (_Float16)(n1 / d1);
      otile[row][lane + 32] = (_Float16)(n2 / d2);
    }
  }
  __syncthreads();

  // ---- Phase 3: y = attn_out @ Wo^T + bo via WMMA ------------------------
  for (int task = wave; task < 32; task += 8) {
    const int mtile = task >> 4;   // 0..1
    const int ntile = task & 15;   // 0..15
    const int n = ntile * 16 + lr;
    v8f c = {};
#pragma unroll
    for (int kk = 0; kk < 2; ++kk) {
      const int arow = mtile * 16 + lr;
      const int akb  = kk * 32 + lh * 8;
      v8h alo = *(const v8h*)&otile[arow][akb];
      v8h ahi = *(const v8h*)&otile[arow][akb + 16];
      const _Float16* wr = Wo16 + (size_t)n * P + kk * 32 + lh * 8;
      v8h blo = *(const v8h*)wr;
      v8h bhi = *(const v8h*)(wr + 16);
      c = __builtin_amdgcn_wmma_f32_16x16x32_f16(false, pack16(alo, ahi),
                                                 false, pack16(blo, bhi),
                                                 (short)0, c, false, false);
    }
    const float bn = bo[n];
#pragma unroll
    for (int r = 0; r < 8; ++r) {
      const int row = b0 + mtile * 16 + r + lh * 8;
      out[(size_t)row * IN + n] = c[r] + bn;
    }
  }
}

extern "C" void kernel_launch(void* const* d_in, const int* in_sizes, int n_in,
                              void* d_out, int out_size, void* d_ws, size_t ws_size,
                              hipStream_t stream) {
  const float* x  = (const float*)d_in[0];
  const float* Wq = (const float*)d_in[1];
  const float* bq = (const float*)d_in[2];
  const float* Wk = (const float*)d_in[3];
  const float* bk = (const float*)d_in[4];
  const float* Wv = (const float*)d_in[5];
  const float* bv = (const float*)d_in[6];
  const float* Wo = (const float*)d_in[7];
  const float* bo = (const float*)d_in[8];

  _Float16* Wqkv16 = (_Float16*)d_ws;
  _Float16* Wo16   = (_Float16*)((char*)d_ws + (size_t)3 * P * IN * sizeof(_Float16));

  const int nB = in_sizes[0] / IN;   // 32768

  convert_weights_kernel<<<(3 * P * IN + 255) / 256, 256, 0, stream>>>(
      Wq, Wk, Wv, Wo, Wqkv16, Wo16);
  fused_attn_kernel<<<nB / TILE_B, 256, 0, stream>>>(
      x, bq, bk, bv, bo, Wqkv16, Wo16, (float*)d_out);
}
